// MoeLayer_24120536334627
// MI455X (gfx1250) — compile-verified, matrix operand order checked by reading, not run on hardware
//
#include <hip/hip_runtime.h>

#define N_TOK 16384
#define DDIM  512
#define FDIM  2048
#define NEXP  8
#define TOPK  2

typedef _Float16 f16;
typedef __attribute__((ext_vector_type(16))) _Float16 v16h;
typedef __attribute__((ext_vector_type(8)))  _Float16 v8h;
typedef __attribute__((ext_vector_type(8)))  float    v8f;
typedef __attribute__((ext_vector_type(4)))  float    v4f;

// ---------------------------------------------------------------------------
// 1) zero output + expert counts
// ---------------------------------------------------------------------------
__global__ void zero_kernel(float* __restrict__ out, int n4, int* __restrict__ counts) {
    int i = blockIdx.x * blockDim.x + threadIdx.x;
    if (i < n4) {
        v4f z = {0.0f, 0.0f, 0.0f, 0.0f};
        ((v4f*)out)[i] = z;
    }
    if (blockIdx.x == 0 && threadIdx.x < NEXP) counts[threadIdx.x] = 0;
}

// ---------------------------------------------------------------------------
// 2) transpose + convert f32 [R][C] -> f16 [C][R], batched over blockIdx.z
// ---------------------------------------------------------------------------
__global__ void transpose_cvt_kernel(const float* __restrict__ src, f16* __restrict__ dst,
                                     int R, int C) {
    __shared__ float tile[32][33];
    const float* s = src + (size_t)blockIdx.z * R * C;
    f16*         d = dst + (size_t)blockIdx.z * R * C;
    int cb = blockIdx.x * 32, rb = blockIdx.y * 32;
    int tx = threadIdx.x, ty = threadIdx.y;   // blockDim = (32, 8)
#pragma unroll
    for (int i = 0; i < 32; i += 8)
        tile[ty + i][tx] = s[(size_t)(rb + ty + i) * C + cb + tx];
    __syncthreads();
#pragma unroll
    for (int i = 0; i < 32; i += 8)
        d[(size_t)(cb + ty + i) * R + rb + tx] = (f16)tile[tx][ty + i];
}

// ---------------------------------------------------------------------------
// 3) gating: one wave32 per token -> top-2 experts + softmax weights
// ---------------------------------------------------------------------------
__global__ void gate_kernel(const float* __restrict__ x, const float* __restrict__ gw,
                            int* __restrict__ topi, float* __restrict__ topw,
                            int* __restrict__ counts) {
    int lane = threadIdx.x & 31;
    int tok  = blockIdx.x * 8 + (threadIdx.x >> 5);
    float acc[NEXP];
#pragma unroll
    for (int e = 0; e < NEXP; e++) acc[e] = 0.0f;
    const float* xr = x + (size_t)tok * DDIM;
    for (int d = lane; d < DDIM; d += 32) {
        float xv = xr[d];
        const float* g = gw + (size_t)d * NEXP;
#pragma unroll
        for (int e = 0; e < NEXP; e++) acc[e] += xv * g[e];
    }
#pragma unroll
    for (int e = 0; e < NEXP; e++) {
#pragma unroll
        for (int off = 16; off > 0; off >>= 1) acc[e] += __shfl_xor(acc[e], off, 32);
    }
    if (lane == 0) {
        int i0 = 0; float v0 = acc[0];
#pragma unroll
        for (int e = 1; e < NEXP; e++) if (acc[e] > v0) { v0 = acc[e]; i0 = e; }
        int i1 = -1; float v1 = -3.4e38f;
#pragma unroll
        for (int e = 0; e < NEXP; e++) if (e != i0 && acc[e] > v1) { v1 = acc[e]; i1 = e; }
        float ex = __expf(v1 - v0);                       // 2-way softmax, numerically safe
        float s  = __builtin_amdgcn_rcpf(1.0f + ex);      // v_rcp_f32, no div chain
        topi[tok * 2 + 0] = i0;  topi[tok * 2 + 1] = i1;
        topw[tok * 2 + 0] = s;   topw[tok * 2 + 1] = ex * s;
        atomicAdd(counts + i0, 1);
        atomicAdd(counts + i1, 1);
    }
}

// ---------------------------------------------------------------------------
// 4) prefix sum over 8 expert counts
// ---------------------------------------------------------------------------
__global__ void offsets_kernel(const int* __restrict__ counts, int* __restrict__ offsets,
                               int* __restrict__ cursor) {
    if (threadIdx.x == 0) {
        int s = 0;
        for (int e = 0; e < NEXP; e++) { offsets[e] = s; cursor[e] = s; s += counts[e]; }
        offsets[NEXP] = s;
    }
}

// ---------------------------------------------------------------------------
// 5) scatter (token, slot) pairs into per-expert contiguous lists
// ---------------------------------------------------------------------------
__global__ void scatter_kernel(const int* __restrict__ topi, const float* __restrict__ topw,
                               int* __restrict__ cursor, int* __restrict__ row_token,
                               float* __restrict__ row_weight) {
    int i = blockIdx.x * blockDim.x + threadIdx.x;
    if (i >= N_TOK * TOPK) return;
    int e   = topi[i];
    int pos = atomicAdd(cursor + e, 1);
    row_token[pos]  = i >> 1;
    row_weight[pos] = topw[i];
}

// ---------------------------------------------------------------------------
// 6) fused expert FFN: out[tok] += w * ( silu(X W1) * (X W3) ) W2
//    block = 256 threads (8 waves), TILE_M = 32 tokens of one expert.
//    Waves: mt = wave>>2 (m-tile 0/1), sub = wave&3 (F n-tile / D d-group).
// ---------------------------------------------------------------------------
__global__ void __launch_bounds__(256)
moe_ffn_kernel(const float* __restrict__ x,
               const f16* __restrict__ w1t,   // [E][F][D] f16
               const f16* __restrict__ w3t,   // [E][F][D] f16
               const f16* __restrict__ w2t,   // [E][D][F] f16
               const int* __restrict__ offsets,
               const int* __restrict__ row_token,
               const float* __restrict__ row_weight,
               float* __restrict__ out) {
    const int e    = blockIdx.y;
    const int base = offsets[e];
    const int cnt  = offsets[e + 1] - base;
    const int m0   = blockIdx.x * 32;
    if (m0 >= cnt) return;

    __shared__ f16   lds_x[32 * DDIM];   // 32 KB: gathered X tile (f16)
    __shared__ f16   lds_h[32 * 64];     //  4 KB: H chunk
    __shared__ int   lds_tok[32];
    __shared__ float lds_w[32];

    const int tid = threadIdx.x;
    if (tid < 32) {
        int r  = m0 + tid;
        int rc = (r < cnt) ? r : (cnt - 1);          // clamp padded rows to a valid token
        lds_tok[tid] = row_token[base + rc];
        lds_w[tid]   = (r < cnt) ? row_weight[base + r] : 0.0f;
    }
    __syncthreads();

    // cooperative gather: 32 rows x 512 cols f32 -> f16 LDS
    {
        int r  = tid >> 3;
        int c0 = (tid & 7) * 64;
        const float* xr = x + (size_t)lds_tok[r] * DDIM + c0;
        f16* dr = lds_x + r * DDIM + c0;
#pragma unroll
        for (int i = 0; i < 64; i += 4) {
            v4f v = *(const v4f*)(xr + i);
            dr[i + 0] = (f16)v[0];
            dr[i + 1] = (f16)v[1];
            dr[i + 2] = (f16)v[2];
            dr[i + 3] = (f16)v[3];
        }
    }
    __syncthreads();

    const int wave  = tid >> 5;
    const int lane  = tid & 31;
    const int lhalf = lane >> 4;     // 0/1
    const int l15   = lane & 15;
    const int kb    = lhalf * 8;     // 16-bit A/B per-lane K base offset

    const int mt  = wave >> 2;       // 0..1: m-tile of 16 rows
    const int sub = wave & 3;        // 0..3: F n-tile (GEMM1) / 128-col d-group (GEMM2)

    const f16* w1e = w1t + (size_t)e * FDIM * DDIM;
    const f16* w3e = w3t + (size_t)e * FDIM * DDIM;
    const f16* w2e = w2t + (size_t)e * DDIM * FDIM;

    v8f acc_o[8];
#pragma unroll
    for (int t = 0; t < 8; t++) acc_o[t] = (v8f){0, 0, 0, 0, 0, 0, 0, 0};

    const int arow = mt * 16 + l15;

    for (int f0 = 0; f0 < FDIM; f0 += 64) {
        // ---- GEMM1: H[32x64] = silu(X W1) * (X W3), one 16x16 tile per wave ----
        v8f a1 = (v8f){0, 0, 0, 0, 0, 0, 0, 0};
        v8f a3 = (v8f){0, 0, 0, 0, 0, 0, 0, 0};
        const int  fcol = f0 + sub * 16 + l15;
        const f16* w1r  = w1e + (size_t)fcol * DDIM;
        const f16* w3r  = w3e + (size_t)fcol * DDIM;
#pragma unroll 4
        for (int ks = 0; ks < DDIM; ks += 32) {
            int k0 = ks + kb;
            v8h alo = *(const v8h*)(lds_x + arow * DDIM + k0);
            v8h ahi = *(const v8h*)(lds_x + arow * DDIM + k0 + 16);
            v16h a = __builtin_shufflevector(alo, ahi, 0,1,2,3,4,5,6,7,8,9,10,11,12,13,14,15);
            v8h b1lo = *(const v8h*)(w1r + k0);
            v8h b1hi = *(const v8h*)(w1r + k0 + 16);
            v16h b1 = __builtin_shufflevector(b1lo, b1hi, 0,1,2,3,4,5,6,7,8,9,10,11,12,13,14,15);
            v8h b3lo = *(const v8h*)(w3r + k0);
            v8h b3hi = *(const v8h*)(w3r + k0 + 16);
            v16h b3 = __builtin_shufflevector(b3lo, b3hi, 0,1,2,3,4,5,6,7,8,9,10,11,12,13,14,15);
            a1 = __builtin_amdgcn_wmma_f32_16x16x32_f16(false, a, false, b1, (short)0, a1, false, false);
            a3 = __builtin_amdgcn_wmma_f32_16x16x32_f16(false, a, false, b3, (short)0, a3, false, false);
        }
        // silu(a1)*a3 -> LDS H.  sigmoid via v_exp + v_rcp (no IEEE div chain).
#pragma unroll
        for (int j = 0; j < 8; j++) {
            float h1  = a1[j];
            float sig = __builtin_amdgcn_rcpf(1.0f + __expf(-h1));
            float h   = h1 * sig * a3[j];
            int   ml  = j + lhalf * 8;
            lds_h[(mt * 16 + ml) * 64 + sub * 16 + l15] = (f16)h;
        }
        __syncthreads();   // H fully written before any wave reads it

        // ---- GEMM2: acc_o += H[mt] @ W2[f0:f0+64, sub*128 : sub*128+128] ----
#pragma unroll
        for (int kk = 0; kk < 64; kk += 32) {
            v8h alo = *(const v8h*)(lds_h + arow * 64 + kk + kb);
            v8h ahi = *(const v8h*)(lds_h + arow * 64 + kk + kb + 16);
            v16h a = __builtin_shufflevector(alo, ahi, 0,1,2,3,4,5,6,7,8,9,10,11,12,13,14,15);
#pragma unroll
            for (int t = 0; t < 8; t++) {
                int dcol = sub * 128 + t * 16 + l15;
                const f16* w2r = w2e + (size_t)dcol * FDIM + f0 + kk + kb;
                v8h blo = *(const v8h*)(w2r);
                v8h bhi = *(const v8h*)(w2r + 16);
                v16h b = __builtin_shufflevector(blo, bhi, 0,1,2,3,4,5,6,7,8,9,10,11,12,13,14,15);
                acc_o[t] = __builtin_amdgcn_wmma_f32_16x16x32_f16(false, a, false, b, (short)0, acc_o[t], false, false);
            }
        }
        __syncthreads();   // done reading H before next chunk overwrites it
    }

    // ---- scale by gate weight, accumulate into out (token hit by 2 experts) ----
#pragma unroll
    for (int t = 0; t < 8; t++) {
        int dcol = sub * 128 + t * 16 + l15;
#pragma unroll
        for (int j = 0; j < 8; j++) {
            int   ml = mt * 16 + j + lhalf * 8;
            float v  = acc_o[t][j] * lds_w[ml];   // padded rows: weight == 0
            atomicAdd(out + (size_t)lds_tok[ml] * DDIM + dcol, v);
        }
    }
}

// ---------------------------------------------------------------------------
// launch
// ---------------------------------------------------------------------------
extern "C" void kernel_launch(void* const* d_in, const int* in_sizes, int n_in,
                              void* d_out, int out_size, void* d_ws, size_t ws_size,
                              hipStream_t stream) {
    (void)in_sizes; (void)n_in; (void)out_size; (void)ws_size;
    const float* x  = (const float*)d_in[0];
    const float* gw = (const float*)d_in[1];
    const float* w1 = (const float*)d_in[2];
    const float* w2 = (const float*)d_in[3];
    const float* w3 = (const float*)d_in[4];
    float* out = (float*)d_out;

    char*  ws  = (char*)d_ws;
    size_t off = 0;
    auto alloc = [&](size_t bytes) -> void* {
        void* p = ws + off;
        off += (bytes + 255) & ~(size_t)255;
        return p;
    };

    const size_t wbytes = (size_t)NEXP * FDIM * DDIM * sizeof(f16);
    f16*   w1t        = (f16*)alloc(wbytes);
    f16*   w3t        = (f16*)alloc(wbytes);
    f16*   w2t        = (f16*)alloc(wbytes);
    int*   topi       = (int*)alloc((size_t)N_TOK * TOPK * sizeof(int));
    float* topw       = (float*)alloc((size_t)N_TOK * TOPK * sizeof(float));
    int*   row_token  = (int*)alloc((size_t)N_TOK * TOPK * sizeof(int));
    float* row_weight = (float*)alloc((size_t)N_TOK * TOPK * sizeof(float));
    int*   counts     = (int*)alloc(64);
    int*   offsets    = (int*)alloc(64);
    int*   cursor     = (int*)alloc(64);

    const int n4 = N_TOK * DDIM / 4;
    zero_kernel<<<(n4 + 255) / 256, 256, 0, stream>>>(out, n4, counts);

    dim3 tb(32, 8);
    transpose_cvt_kernel<<<dim3(FDIM / 32, DDIM / 32, NEXP), tb, 0, stream>>>(w1, w1t, DDIM, FDIM);
    transpose_cvt_kernel<<<dim3(FDIM / 32, DDIM / 32, NEXP), tb, 0, stream>>>(w3, w3t, DDIM, FDIM);
    transpose_cvt_kernel<<<dim3(DDIM / 32, FDIM / 32, NEXP), tb, 0, stream>>>(w2, w2t, FDIM, DDIM);

    gate_kernel<<<N_TOK / 8, 256, 0, stream>>>(x, gw, topi, topw, counts);
    offsets_kernel<<<1, 32, 0, stream>>>(counts, offsets, cursor);
    scatter_kernel<<<(N_TOK * TOPK + 255) / 256, 256, 0, stream>>>(topi, topw, cursor,
                                                                   row_token, row_weight);

    moe_ffn_kernel<<<dim3((N_TOK * TOPK + 31) / 32, NEXP), 256, 0, stream>>>(
        x, w1t, w3t, w2t, offsets, row_token, row_weight, out);
}
